// MTREncoder_12962211299520
// MI455X (gfx1250) — matrix operand order repaired and checked by reading, not compiled, not run on hardware
//
#include <hip/hip_runtime.h>
#include <hip/hip_bf16.h>

// ---------------------------------------------------------------------------
// MTR encoder for gfx1250 (MI455X). Dense GEMMs: v_wmma_f32_16x16x32_bf16,
// 128x64 block tiles, 8 waves, 4 WMMAs per wave per K-step. LDS tiles are
// stored pre-swizzled into WMMA fragment layout, so fragment loads are
// contiguous 32B bf16x16 loads (ds_load_b128 pairs) with no repacking.
// ---------------------------------------------------------------------------

typedef __bf16 bf16_t;
typedef bf16_t bf16x16 __attribute__((ext_vector_type(16)));
typedef float  f32x8   __attribute__((ext_vector_type(8)));

// ===========================================================================
// GEMM: C[M,N] = act( A[M,K] @ W[K,N] * scale + bias )
// Requires: K%32==0, N%64==0, M%128==0 (guaranteed by padding below).
// Block = 256 threads = 8 waves; block tile 128(M) x 64(N); wave w owns rows
// [w*16, w*16+16) x all 64 cols -> 4 accumulators, 4 WMMAs per K-step.
// LDS layout = fragment layout: sA[(w*32+lane)*16 + e], sB[(t*32+lane)*16 + i].
// ===========================================================================
__global__ __launch_bounds__(256)
void gemm_bf16_wmma(const float* __restrict__ A, const float* __restrict__ W,
                    const float* __restrict__ bias, float* __restrict__ C,
                    int M, int K, int N, float scale, int do_relu)
{
    alignas(32) __shared__ bf16_t sA[128 * 32];   // 8 KB, fragment-ordered
    alignas(32) __shared__ bf16_t sB[32 * 64];    // 4 KB, fragment-ordered

    const int tid  = threadIdx.x;
    const int lane = tid & 31;
    const int wv   = tid >> 5;            // 0..7
    const int m0   = blockIdx.x * 128;
    const int n0   = blockIdx.y * 64;

    f32x8 acc[4];
#pragma unroll
    for (int i = 0; i < 4; ++i)
        acc[i] = (f32x8){0.f, 0.f, 0.f, 0.f, 0.f, 0.f, 0.f, 0.f};

    for (int k0 = 0; k0 < K; k0 += 32) {
        // Speculative prefetch of the next K tile (global_prefetch_b8).
        {
            int kn = (k0 + 32 < K) ? (k0 + 32) : k0;
            __builtin_prefetch(A + (size_t)(m0 + (tid >> 1)) * K + kn + (tid & 1) * 16, 0, 0);
            __builtin_prefetch(W + (size_t)(kn + (tid & 31)) * N + n0 + (tid >> 5) * 8, 0, 0);
        }
        // ---- Stage A: 128x32 f32 -> bf16, written in A-fragment order. ----
        // float4 chunk (row r, K c4..c4+3) -> lane lsel, elements e0..e0+3.
#pragma unroll
        for (int it = 0; it < 4; ++it) {
            int i  = tid + it * 256;            // 0..1023
            int r  = i >> 3;                    // 0..127
            int c4 = (i & 7) << 2;              // 0,4,...,28
            const float4 f = *(const float4*)(A + (size_t)(m0 + r) * K + k0 + c4);
            int w    = r >> 4;
            int lsel = (r & 15) + (((c4 >> 3) & 1) << 4);
            int e0   = ((c4 >> 4) << 3) + (c4 & 4);
            union { bf16_t h[4]; unsigned long long u; } pk;
            pk.h[0] = (bf16_t)f.x; pk.h[1] = (bf16_t)f.y;
            pk.h[2] = (bf16_t)f.z; pk.h[3] = (bf16_t)f.w;
            *(unsigned long long*)&sA[((w * 32 + lsel) << 4) + e0] = pk.u;
        }
        // ---- Stage B: 32x64 f32 -> bf16, written in B-fragment order. ----
        // Column of 4 K (kk..kk+3) at col n -> lane lb, elements i0..i0+3.
#pragma unroll
        for (int it = 0; it < 2; ++it) {
            int i  = tid + it * 256;            // 0..511
            int q  = i >> 6;                    // 0..7
            int n  = i & 63;
            int kk = q << 2;
            const float* wp = W + (size_t)(k0 + kk) * N + n0 + n;
            union { bf16_t h[4]; unsigned long long u; } pk;
            pk.h[0] = (bf16_t)wp[0];
            pk.h[1] = (bf16_t)wp[(size_t)N];
            pk.h[2] = (bf16_t)wp[2 * (size_t)N];
            pk.h[3] = (bf16_t)wp[3 * (size_t)N];
            int t  = n >> 4;
            int lb = (((kk >> 3) & 1) << 4) + (n & 15);
            int i0 = ((kk >> 4) << 3) + (kk & 4);
            *(unsigned long long*)&sB[((t * 32 + lb) << 4) + i0] = pk.u;
        }
        __syncthreads();

        // ---- Fragment loads: contiguous 32B per lane (2x ds_load_b128). ----
        const bf16x16 af = *(const bf16x16*)&sA[(wv * 32 + lane) << 4];
        bf16x16 bfr[4];
#pragma unroll
        for (int t = 0; t < 4; ++t)
            bfr[t] = *(const bf16x16*)&sB[(t * 32 + lane) << 4];

#pragma unroll
        for (int t = 0; t < 4; ++t)
            acc[t] = __builtin_amdgcn_wmma_f32_16x16x32_bf16(
                false, af, false, bfr[t], (short)0, acc[t], false, false);
        __syncthreads();
    }

    // ---- Epilogue per C/D layout: VGPR r -> row rbase+r, col lane&15. ----
    const int cb    = lane & 15;
    const int rbase = m0 + wv * 16 + ((lane >> 4) << 3);
#pragma unroll
    for (int t = 0; t < 4; ++t) {
        int ncol = n0 + t * 16 + cb;
        float bv = bias ? bias[ncol] : 0.f;
#pragma unroll
        for (int r = 0; r < 8; ++r) {
            float v = acc[t][r] * scale + bv;
            if (do_relu) v = v > 0.f ? v : 0.f;
            C[(size_t)(rbase + r) * N + ncol] = v;
        }
    }
}

// ===========================================================================
// Padding helpers (make every GEMM operand K%32==0 without hot-loop guards)
// ===========================================================================
__global__ void pad_cols_kernel(const float* __restrict__ src, float* __restrict__ dst,
                                int R, int K, int Kp)
{
    int idx = blockIdx.x * blockDim.x + threadIdx.x;
    if (idx >= R * Kp) return;
    int r = idx / Kp, c = idx - r * Kp;
    dst[idx] = (c < K) ? src[(size_t)r * K + c] : 0.f;
}

__global__ void pad_rows_kernel(const float* __restrict__ src, float* __restrict__ dst,
                                int K, int Kp, int N)
{
    int idx = blockIdx.x * blockDim.x + threadIdx.x;
    if (idx >= Kp * N) return;
    int r = idx / N;
    dst[idx] = (r < K) ? src[idx] : 0.f;
}

// obj_in[r, 0:29]=traj, [29]=mask, [30:32]=0  (K padded to 32)
__global__ void build_obj_in_kernel(const float* __restrict__ traj,
                                    const int* __restrict__ m,
                                    float* __restrict__ out, int rows)
{
    int idx = blockIdx.x * blockDim.x + threadIdx.x;
    if (idx >= rows * 32) return;
    int r = idx >> 5, c = idx & 31;
    float v = 0.f;
    if (c < 29)       v = traj[(size_t)r * 29 + c];
    else if (c == 29) v = m[r] ? 1.f : 0.f;
    out[idx] = v;
}

// ===========================================================================
// PointNet pooling
// ===========================================================================
__global__ void pool_concat_kernel(const float* __restrict__ f,
                                   const int* __restrict__ m,
                                   float* __restrict__ cat, int T, int C)
{
    int bn = blockIdx.x;
    for (int c = threadIdx.x; c < C; c += blockDim.x) {
        float mx = -3.4e38f;
        for (int t = 0; t < T; ++t) {
            float v = m[bn * T + t] ? f[((size_t)bn * T + t) * C + c] : 0.f;
            mx = fmaxf(mx, v);
        }
        for (int t = 0; t < T; ++t) {
            float v = m[bn * T + t] ? f[((size_t)bn * T + t) * C + c] : 0.f;
            cat[((size_t)bn * T + t) * (2 * C) + c]     = v;
            cat[((size_t)bn * T + t) * (2 * C) + C + c] = mx;
        }
    }
}

__global__ void pool_final_kernel(const float* __restrict__ f,
                                  const int* __restrict__ m,
                                  float* __restrict__ g, int* __restrict__ valid,
                                  int T, int C)
{
    int bn = blockIdx.x;
    int anyv = 0;
    for (int t = 0; t < T; ++t) anyv |= (m[bn * T + t] != 0);
    if (threadIdx.x == 0) valid[bn] = anyv;
    for (int c = threadIdx.x; c < C; c += blockDim.x) {
        float mx = -3.4e38f;
        for (int t = 0; t < T; ++t) {
            float v = m[bn * T + t] ? f[((size_t)bn * T + t) * C + c] : 0.f;
            mx = fmaxf(mx, v);
        }
        g[(size_t)bn * C + c] = mx;
    }
}

// ===========================================================================
// Token assembly / PE / KNN
// ===========================================================================
__global__ void scatter_feat_kernel(const float* __restrict__ objf,
                                    const int* __restrict__ objv,
                                    const float* __restrict__ mapf,
                                    const int* __restrict__ mapv,
                                    const float* __restrict__ objpos,
                                    const float* __restrict__ mappos,
                                    float* __restrict__ feat,
                                    float* __restrict__ pos,
                                    int* __restrict__ mask)
{
    int row = blockIdx.x;            // b*832+n
    int b = row / 832, n = row - b * 832;
    const float* srcf; const float* srcp; int vld;
    if (n < 64) {
        int si = b * 64 + n;
        srcf = objf + (size_t)si * 256; srcp = objpos + (size_t)si * 3; vld = objv[si];
    } else {
        int si = b * 768 + (n - 64);
        srcf = mapf + (size_t)si * 256; srcp = mappos + (size_t)si * 3; vld = mapv[si];
    }
    float fm = vld ? 1.f : 0.f;
    for (int c = threadIdx.x; c < 256; c += blockDim.x)
        feat[(size_t)row * 256 + c] = srcf[c] * fm;
    if (threadIdx.x < 3) pos[row * 3 + threadIdx.x] = srcp[threadIdx.x];
    if (threadIdx.x == 0) mask[row] = vld;
}

__global__ void pe_kernel(const float* __restrict__ pos, float* __restrict__ pe)
{
    int row = blockIdx.x;
    float x = pos[row * 3 + 0], y = pos[row * 3 + 1];
    const float TWO_PI = 6.283185307179586f;
    for (int c = threadIdx.x; c < 256; c += blockDim.x) {
        int e = c & 127;
        float src = (c < 128) ? y : x;
        float dim = powf(10000.f, (float)(2 * (e >> 1)) * (1.f / 128.f));
        float arg = src * TWO_PI / dim;
        pe[(size_t)row * 256 + c] = (e & 1) ? cosf(arg) : sinf(arg);
    }
}

__global__ void knn_kernel(const float* __restrict__ pos,
                           const int* __restrict__ mask,
                           int* __restrict__ nbi, int* __restrict__ nbv,
                           int total, int NT)
{
    int row = blockIdx.x * blockDim.x + threadIdx.x;
    if (row >= total) return;
    int b = row / NT;
    int base = b * NT;
    float px = pos[row * 3], py = pos[row * 3 + 1], pz = pos[row * 3 + 2];
    float bd[16]; int bi[16];
#pragma unroll
    for (int i = 0; i < 16; ++i) { bd[i] = 3.4e38f; bi[i] = 0; }
    for (int j = 0; j < NT; ++j) {
        float d;
        if (mask[base + j]) {
            float dx = px - pos[(base + j) * 3];
            float dy = py - pos[(base + j) * 3 + 1];
            float dz = pz - pos[(base + j) * 3 + 2];
            d = dx * dx + dy * dy + dz * dz;
        } else {
            d = 1e30f;       // masked: stable fill after all valid entries
        }
        if (d < bd[15]) {
            int p = 15;
            while (p > 0 && d < bd[p - 1]) { bd[p] = bd[p - 1]; bi[p] = bi[p - 1]; --p; }
            bd[p] = d; bi[p] = j;
        }
    }
    for (int t = 0; t < 16; ++t) {
        nbi[row * 16 + t] = bi[t];
        nbv[row * 16 + t] = mask[base + bi[t]];
    }
}

__global__ void add_kernel(const float* __restrict__ a, const float* __restrict__ b,
                           float* __restrict__ o, int n)
{
    int i = blockIdx.x * blockDim.x + threadIdx.x;
    if (i < n) o[i] = a[i] + b[i];
}

// Gathered local attention: one wave per (token,head); lane = channel (dh=32).
__global__ __launch_bounds__(256)
void attn_kernel(const float* __restrict__ q, const float* __restrict__ k,
                 const float* __restrict__ v, const int* __restrict__ nbi,
                 const int* __restrict__ nbv, float* __restrict__ ao,
                 int NT, float scl)
{
    int row  = blockIdx.x;
    int h    = threadIdx.x >> 5;
    int lane = threadIdx.x & 31;
    int b    = row / NT;
    const float qv = q[(size_t)row * 256 + h * 32 + lane];
    int jr[16];
    float sc[16];
#pragma unroll
    for (int t = 0; t < 16; ++t) {
        int jrow = b * NT + nbi[row * 16 + t];
        jr[t] = jrow;
        float p = qv * k[(size_t)jrow * 256 + h * 32 + lane];
        for (int off = 16; off > 0; off >>= 1) p += __shfl_xor(p, off, 32);
        sc[t] = nbv[row * 16 + t] ? p * scl : -1e9f;
    }
    float mx = sc[0];
#pragma unroll
    for (int t = 1; t < 16; ++t) mx = fmaxf(mx, sc[t]);
    float s = 0.f;
#pragma unroll
    for (int t = 0; t < 16; ++t) { sc[t] = expf(sc[t] - mx); s += sc[t]; }
    float inv = 1.f / s;
    float outv = 0.f;
#pragma unroll
    for (int t = 0; t < 16; ++t)
        outv += sc[t] * v[(size_t)jr[t] * 256 + h * 32 + lane];
    ao[(size_t)row * 256 + h * 32 + lane] = outv * inv;
}

// out = LayerNorm(x + r) * g + be ; one wave per row, 8 channels per lane.
__global__ __launch_bounds__(256)
void add_ln_kernel(const float* __restrict__ x, const float* __restrict__ r,
                   const float* __restrict__ g, const float* __restrict__ be,
                   float* __restrict__ out, int rows)
{
    int wv = threadIdx.x >> 5;
    int lane = threadIdx.x & 31;
    int row = blockIdx.x * 8 + wv;
    if (row >= rows) return;
    float vals[8];
    float s = 0.f;
#pragma unroll
    for (int i = 0; i < 8; ++i) {
        int c = lane + i * 32;
        vals[i] = x[(size_t)row * 256 + c] + r[(size_t)row * 256 + c];
        s += vals[i];
    }
    for (int off = 16; off > 0; off >>= 1) s += __shfl_xor(s, off, 32);
    float mu = s * (1.f / 256.f);
    float vs = 0.f;
#pragma unroll
    for (int i = 0; i < 8; ++i) { float d = vals[i] - mu; vs += d * d; }
    for (int off = 16; off > 0; off >>= 1) vs += __shfl_xor(vs, off, 32);
    float inv = rsqrtf(vs * (1.f / 256.f) + 1e-5f);
#pragma unroll
    for (int i = 0; i < 8; ++i) {
        int c = lane + i * 32;
        out[(size_t)row * 256 + c] = (vals[i] - mu) * inv * g[c] + be[c];
    }
}

__global__ void write_out_kernel(const float* __restrict__ feat,
                                 const int* __restrict__ mask,
                                 const int* __restrict__ track,
                                 float* __restrict__ out)
{
    int row = blockIdx.x;
    int b = row / 832, n = row - b * 832;
    float fm = mask[row] ? 1.f : 0.f;
    float* dst;
    if (n < 64) dst = out + 4096 + ((size_t)b * 64 + n) * 256;
    else        dst = out + 4096 + 262144 + ((size_t)b * 768 + (n - 64)) * 256;
    bool isCenter = (n < 64) && (track[b] == n);
    for (int c = threadIdx.x; c < 256; c += blockDim.x) {
        float v = feat[(size_t)row * 256 + c] * fm;
        dst[c] = v;
        if (isCenter) out[(size_t)b * 256 + c] = v;
    }
}

// ===========================================================================
// Host orchestration
// ===========================================================================
static inline void launch_gemm(const float* A, const float* W, const float* bias,
                               float* C, int M, int K, int N,
                               float scale, int relu, hipStream_t s)
{
    dim3 grid(M / 128, N / 64);
    gemm_bf16_wmma<<<grid, 256, 0, s>>>(A, W, bias, C, M, K, N, scale, relu);
}

extern "C" void kernel_launch(void* const* d_in, const int* in_sizes, int n_in,
                              void* d_out, int out_size, void* d_ws, size_t ws_size,
                              hipStream_t stream)
{
    (void)in_sizes; (void)n_in; (void)out_size; (void)ws_size;

    const float* obj_trajs = (const float*)d_in[0];
    const int*   obj_mask  = (const int*)  d_in[1];
    const float* map_poly  = (const float*)d_in[2];
    const int*   map_mask  = (const int*)  d_in[3];
    const float* obj_pos   = (const float*)d_in[4];
    const float* map_pos   = (const float*)d_in[5];
    const int*   track     = (const int*)  d_in[6];
    const float* ag_pre_w0 = (const float*)d_in[7];
    const float* ag_mlp_w0 = (const float*)d_in[8];
    const float* ag_mlp_w1 = (const float*)d_in[9];
    const float* ag_out_w0 = (const float*)d_in[10];
    const float* ag_out_b0 = (const float*)d_in[11];
    const float* ag_out_w1 = (const float*)d_in[12];
    const float* ag_out_b1 = (const float*)d_in[13];
    const float* mp_pre_w0 = (const float*)d_in[14];
    const float* mp_pre_w1 = (const float*)d_in[15];
    const float* mp_pre_w2 = (const float*)d_in[16];
    const float* mp_mlp_w0 = (const float*)d_in[17];
    const float* mp_mlp_w1 = (const float*)d_in[18];
    const float* mp_out_w0 = (const float*)d_in[19];
    const float* mp_out_b0 = (const float*)d_in[20];
    const float* mp_out_w1 = (const float*)d_in[21];
    const float* mp_out_b1 = (const float*)d_in[22];
    const float* Wq = (const float*)d_in[23]; const float* bq = (const float*)d_in[24];
    const float* Wk = (const float*)d_in[25]; const float* bk = (const float*)d_in[26];
    const float* Wv = (const float*)d_in[27]; const float* bv = (const float*)d_in[28];
    const float* Wo = (const float*)d_in[29]; const float* bo = (const float*)d_in[30];
    const float* W1 = (const float*)d_in[31]; const float* b1 = (const float*)d_in[32];
    const float* W2 = (const float*)d_in[33]; const float* b2 = (const float*)d_in[34];
    const float* ln1_g = (const float*)d_in[35]; const float* ln1_b = (const float*)d_in[36];
    const float* ln2_g = (const float*)d_in[37]; const float* ln2_b = (const float*)d_in[38];

    float* out = (float*)d_out;
    char*  ws  = (char*)d_ws;

    const int Bn = 16, NO = 64, Ttr = 11, P = 768, PT = 20;
    const int D = 256, HM = 64, Ntok = 832, FF = 1024, Lyr = 6;
    const int ROWS = Bn * Ntok;                     // 13312
    const float BN_SCALE = 0.9999950000374997f;
    const float ATTN_SCALE = 0.17677669529663687f;  // 1/sqrt(32)

    // ---- workspace layout (bytes) -----------------------------------------
    const size_t SZ_ROWD = (size_t)ROWS * D * 4;    // 13,631,488
    size_t OFF_FEAT = 0;
    size_t OFF_PE   = OFF_FEAT + SZ_ROWD;
    size_t OFF_POS  = OFF_PE   + SZ_ROWD;
    size_t OFF_MASK = OFF_POS  + (size_t)ROWS * 3 * 4;
    size_t OFF_NBI  = OFF_MASK + (size_t)ROWS * 4;
    size_t OFF_NBV  = OFF_NBI  + (size_t)ROWS * 16 * 4;
    size_t OFF_OBJF = OFF_NBV  + (size_t)ROWS * 16 * 4;
    size_t OFF_OBJV = OFF_OBJF + (size_t)Bn * NO * D * 4;
    size_t S        = OFF_OBJV + (size_t)Bn * NO * 4;   // aliased scratch base
    S = (S + 255) & ~(size_t)255;

    float* feat   = (float*)(ws + OFF_FEAT);
    float* pe     = (float*)(ws + OFF_PE);
    float* pos    = (float*)(ws + OFF_POS);
    int*   mask832= (int*)  (ws + OFF_MASK);
    int*   nbi    = (int*)  (ws + OFF_NBI);
    int*   nbv    = (int*)  (ws + OFF_NBV);
    float* objf   = (float*)(ws + OFF_OBJF);
    int*   objv   = (int*)  (ws + OFF_OBJV);

    // ======================= Agent PointNet (scratch S) ====================
    {
        int rows = Bn * NO * Ttr;                       // 11264
        float* obj_in = (float*)(ws + S);                               // [rows,32]
        float* f1     = (float*)(ws + S + (size_t)16 * 1024 * 1024);    // [rows,256]
        float* cat    = (float*)(ws + S + (size_t)32 * 1024 * 1024);    // [rows,512]
        float* f2     = (float*)(ws + S + (size_t)64 * 1024 * 1024);
        float* f3     = (float*)(ws + S + (size_t)80 * 1024 * 1024);
        float* gA     = (float*)(ws + S + (size_t)96 * 1024 * 1024);    // [1024,256]
        float* hA     = (float*)(ws + S + (size_t)98 * 1024 * 1024);
        float* wpadA  = (float*)(ws + S + (size_t)100 * 1024 * 1024);   // [32,256]

        build_obj_in_kernel<<<(rows * 32 + 255) / 256, 256, 0, stream>>>(
            obj_trajs, obj_mask, obj_in, rows);
        pad_rows_kernel<<<(32 * D + 255) / 256, 256, 0, stream>>>(
            ag_pre_w0, wpadA, 30, 32, D);
        launch_gemm(obj_in, wpadA, nullptr, f1, rows, 32, D, BN_SCALE, 1, stream);
        pool_concat_kernel<<<Bn * NO, 256, 0, stream>>>(f1, obj_mask, cat, Ttr, D);
        launch_gemm(cat, ag_mlp_w0, nullptr, f2, rows, 2 * D, D, BN_SCALE, 1, stream);
        launch_gemm(f2, ag_mlp_w1, nullptr, f3, rows, D, D, BN_SCALE, 1, stream);
        pool_final_kernel<<<Bn * NO, 256, 0, stream>>>(f3, obj_mask, gA, objv, Ttr, D);
        launch_gemm(gA, ag_out_w0, ag_out_b0, hA, Bn * NO, D, D, 1.f, 1, stream);
        launch_gemm(hA, ag_out_w1, ag_out_b1, objf, Bn * NO, D, D, 1.f, 0, stream);
    }

    // ======================= Map PointNet (scratch S) ======================
    float* mapf; int* mapv;
    {
        int rows = Bn * P * PT;                         // 245760
        size_t SZ64 = (size_t)rows * HM * 4;            // 62.9MB
        float* bufA = (float*)(ws + S);
        float* bufB = (float*)(ws + S + SZ64);
        float* bufC = (float*)(ws + S + 2 * SZ64);      // [rows,128]; aliases mpad
        float* mpad = bufC;                             // [rows,32] padded input
        size_t TAIL = S + 4 * SZ64;
        float* wpadM = (float*)(ws + TAIL);                            // [32,64]
        float* gM    = (float*)(ws + TAIL + (size_t)1 * 1024 * 1024);
        mapv         = (int*)  (ws + TAIL + (size_t)5 * 1024 * 1024);
        float* hM    = (float*)(ws + TAIL + (size_t)6 * 1024 * 1024);
        mapf         = (float*)(ws + TAIL + (size_t)10 * 1024 * 1024); // [12288,256]

        pad_cols_kernel<<<(rows * 32 + 255) / 256, 256, 0, stream>>>(
            map_poly, mpad, rows, 9, 32);
        pad_rows_kernel<<<(32 * HM + 255) / 256, 256, 0, stream>>>(
            mp_pre_w0, wpadM, 9, 32, HM);
        launch_gemm(mpad, wpadM, nullptr, bufA, rows, 32, HM, BN_SCALE, 1, stream);
        launch_gemm(bufA, mp_pre_w1, nullptr, bufB, rows, HM, HM, BN_SCALE, 1, stream);
        launch_gemm(bufB, mp_pre_w2, nullptr, bufA, rows, HM, HM, BN_SCALE, 1, stream);
        pool_concat_kernel<<<Bn * P, 128, 0, stream>>>(bufA, map_mask, bufC, PT, HM);
        launch_gemm(bufC, mp_mlp_w0, nullptr, bufB, rows, 2 * HM, HM, BN_SCALE, 1, stream);
        launch_gemm(bufB, mp_mlp_w1, nullptr, bufA, rows, HM, HM, BN_SCALE, 1, stream);
        pool_final_kernel<<<Bn * P, 128, 0, stream>>>(bufA, map_mask, gM, mapv, PT, HM);
        launch_gemm(gM, mp_out_w0, mp_out_b0, hM, Bn * P, HM, HM, 1.f, 1, stream);
        launch_gemm(hM, mp_out_w1, mp_out_b1, mapf, Bn * P, HM, D, 1.f, 0, stream);
    }

    // ======================= Token assembly, PE, KNN =======================
    scatter_feat_kernel<<<ROWS, 256, 0, stream>>>(objf, objv, mapf, mapv,
                                                  obj_pos, map_pos,
                                                  feat, pos, mask832);
    pe_kernel<<<ROWS, 256, 0, stream>>>(pos, pe);
    knn_kernel<<<(ROWS + 255) / 256, 256, 0, stream>>>(pos, mask832, nbi, nbv,
                                                       ROWS, Ntok);

    // ======================= Transformer stack (scratch S) =================
    float* qkin  = (float*)(ws + S + 0 * SZ_ROWD);
    float* qb    = (float*)(ws + S + 1 * SZ_ROWD);
    float* kb_   = (float*)(ws + S + 2 * SZ_ROWD);
    float* vb    = (float*)(ws + S + 3 * SZ_ROWD);
    float* aob   = (float*)(ws + S + 4 * SZ_ROWD);
    float* proj  = (float*)(ws + S + 5 * SZ_ROWD);
    float* feat2 = (float*)(ws + S + 6 * SZ_ROWD);
    float* ffh   = (float*)(ws + S + 7 * SZ_ROWD);                     // [ROWS,1024]
    float* ff2   = (float*)(ws + S + 7 * SZ_ROWD + (size_t)ROWS * FF * 4);

    const int nElem = ROWS * D;
    for (int l = 0; l < Lyr; ++l) {
        const float* Wql = Wq + (size_t)l * D * D; const float* bql = bq + (size_t)l * D;
        const float* Wkl = Wk + (size_t)l * D * D; const float* bkl = bk + (size_t)l * D;
        const float* Wvl = Wv + (size_t)l * D * D; const float* bvl = bv + (size_t)l * D;
        const float* Wol = Wo + (size_t)l * D * D; const float* bol = bo + (size_t)l * D;
        const float* W1l = W1 + (size_t)l * D * FF; const float* b1l = b1 + (size_t)l * FF;
        const float* W2l = W2 + (size_t)l * FF * D; const float* b2l = b2 + (size_t)l * D;

        add_kernel<<<(nElem + 255) / 256, 256, 0, stream>>>(feat, pe, qkin, nElem);
        launch_gemm(qkin, Wql, bql, qb,  ROWS, D, D, 1.f, 0, stream);
        launch_gemm(qkin, Wkl, bkl, kb_, ROWS, D, D, 1.f, 0, stream);
        launch_gemm(feat, Wvl, bvl, vb,  ROWS, D, D, 1.f, 0, stream);
        attn_kernel<<<ROWS, 256, 0, stream>>>(qb, kb_, vb, nbi, nbv, aob,
                                              Ntok, ATTN_SCALE);
        launch_gemm(aob, Wol, bol, proj, ROWS, D, D, 1.f, 0, stream);
        add_ln_kernel<<<(ROWS + 7) / 8, 256, 0, stream>>>(
            feat, proj, ln1_g + (size_t)l * D, ln1_b + (size_t)l * D, feat2, ROWS);
        launch_gemm(feat2, W1l, b1l, ffh, ROWS, D, FF, 1.f, 1, stream);
        launch_gemm(ffh, W2l, b2l, ff2, ROWS, FF, D, 1.f, 0, stream);
        add_ln_kernel<<<(ROWS + 7) / 8, 256, 0, stream>>>(
            feat2, ff2, ln2_g + (size_t)l * D, ln2_b + (size_t)l * D, feat, ROWS);
    }

    // ======================= Outputs =======================================
    write_out_kernel<<<ROWS, 256, 0, stream>>>(feat, mask832, track, out);
}
